// FeatureNetGCN_57964878627087
// MI455X (gfx1250) — matrix operand order repaired and checked by reading, not compiled, not run on hardware
//
#include <hip/hip_runtime.h>

// ---------------------------------------------------------------- constants
#define Bc   16
#define Nc   4096
#define Dc   128
#define Kc   16
#define BNc  (Bc * Nc)
#define ROWS 32              // i-rows per kNN block (2 WMMA sub-tiles)
#define STRIP 256            // kNN distance strip width (cols) staged in LDS
#define NSTRIP (Nc / STRIP)  // 16 strips

// ---------------------------------------------------------------- WMMA types
typedef __attribute__((ext_vector_type(16))) __bf16 v16bf;
typedef __attribute__((ext_vector_type(8)))  float  v8f;
typedef __attribute__((ext_vector_type(4)))  unsigned u32x4;

union V16 { v16bf v; u32x4 q[2]; unsigned u[8]; };

// optional gfx1250 async global->LDS path (probe via __has_builtin)
#if defined(__has_builtin)
#if __has_builtin(__builtin_amdgcn_global_load_async_to_lds_b128)
#define HAS_ASYNC_LDS 1
#endif
#endif
#ifndef HAS_ASYNC_LDS
#define HAS_ASYNC_LDS 0
#endif

#if HAS_ASYNC_LDS
// exact param types per clang diagnostic: 'int __vector(4) __device__ *' (AS1)
// and the LDS side in AS3 ('__shared__').
typedef int v4i_vs __attribute__((vector_size(16)));
typedef __attribute__((address_space(1))) v4i_vs gvec_t;
typedef __attribute__((address_space(3))) v4i_vs lvec_t;
#endif

// f32 -> bf16 (round to nearest even), bit-level to avoid header classes
__device__ __forceinline__ unsigned short f2bf(float f) {
    unsigned u = __float_as_uint(f);
    unsigned r = (u + 0x7FFFu + ((u >> 16) & 1u)) >> 16;
    return (unsigned short)r;
}

// ---------------------------------------------------------------- prep kernels
__global__ __launch_bounds__(128) void k_prep1(const float* __restrict__ x,
                                               unsigned short* __restrict__ hbf,
                                               float* __restrict__ sq) {
    __shared__ float red[128];
    const int node = blockIdx.x;
    const int b = node >> 12, n = node & (Nc - 1);
    const int c = threadIdx.x;
    float v = (c < 3) ? x[((size_t)b * 3 + c) * Nc + n] : 0.0f;
    hbf[(size_t)node * Dc + c] = f2bf(v);
    red[c] = v * v;
    __syncthreads();
    for (int s = 64; s > 0; s >>= 1) {
        if (c < s) red[c] += red[c + s];
        __syncthreads();
    }
    if (c == 0) sq[node] = red[0];
}

__global__ __launch_bounds__(128) void k_prep(const float* __restrict__ h,
                                              unsigned short* __restrict__ hbf,
                                              float* __restrict__ sq) {
    __shared__ float red[128];
    const int node = blockIdx.x;
    const int c = threadIdx.x;
    float v = h[(size_t)node * Dc + c];
    hbf[(size_t)node * Dc + c] = f2bf(v);
    red[c] = v * v;
    __syncthreads();
    for (int s = 64; s > 0; s >>= 1) {
        if (c < s) red[c] += red[c + s];
        __syncthreads();
    }
    if (c == 0) sq[node] = red[0];
}

// W (dIn x 128) f32 -> Wt bf16 (128 x wrow), transposed + zero padded
__global__ void k_castW(const float* __restrict__ W, unsigned short* __restrict__ wbf,
                        int dIn, int wrow) {
    int idx = blockIdx.x * blockDim.x + threadIdx.x;
    if (idx >= Dc * wrow) return;
    int cc = idx / wrow;
    int k  = idx % wrow;
    float v = (k < dIn) ? W[(size_t)k * Dc + cc] : 0.0f;
    wbf[(size_t)cc * wrow + k] = f2bf(v);
}

__global__ void k_zero(float* __restrict__ p, int n) {
    int i = blockIdx.x * blockDim.x + threadIdx.x;
    int stride = gridDim.x * blockDim.x;
    for (; i < n; i += stride) p[i] = 0.0f;
}

// ---------------------------------------------------------------- kNN (WMMA Gram)
// Block = 256 thr (8 waves) handles ROWS=32 i-rows: two resident A sub-tiles per
// wave, each loaded B-tile feeds 2*KT WMMAs. Distances staged per 256-col strip
// in LDS; 8 threads/row maintain running top-16; LDS aliased for the final merge.
template <int KT>
__global__ __launch_bounds__(256) void k_knn(const unsigned short* __restrict__ hbf,
                                             const float* __restrict__ sq,
                                             int* __restrict__ nbr) {
    __shared__ unsigned smem[ROWS * STRIP];   // 32 KB: dist strip, then cand merge
    __shared__ float s_sqi[ROWS];
    float* s_dist = (float*)smem;             // [ROWS][STRIP]
    float* s_cd   = (float*)smem;             // [ROWS][128]  (aliased)
    int*   s_ci   = (int*)(smem + ROWS * 128);// [ROWS][128]  (aliased)

    const int tid  = threadIdx.x;
    const int lane = tid & 31;
    const int wv   = tid >> 5;
    const int b    = blockIdx.x >> 7;         // N/ROWS = 128 i-tiles per batch
    const int it   = blockIdx.x & 127;
    const int i0   = it * ROWS;
    const size_t baseRow = (size_t)b * Nc;

    if (tid < ROWS) s_sqi[tid] = sq[baseRow + i0 + tid];

    const int m  = lane & 15;
    const int kh = lane >> 4;

    // two A sub-tiles (rows i0+m, i0+16+m), ISA 16-bit A 16x32 layout, b128 loads
    V16 a[2][KT];
#pragma unroll
    for (int s = 0; s < 2; ++s) {
        const unsigned short* row = hbf + (baseRow + i0 + s * 16 + m) * Dc;
#pragma unroll
        for (int kt = 0; kt < KT; ++kt) {
            a[s][kt].q[0] = *(const u32x4*)(row + kt * 32 + kh * 8);
            a[s][kt].q[1] = *(const u32x4*)(row + kt * 32 + 16 + kh * 8);
        }
    }

    float bd[16];
    int   bi[16];
#pragma unroll
    for (int s = 0; s < 16; ++s) { bd[s] = __builtin_inff(); bi[s] = 0; }
    const int r_row = tid >> 3;   // 0..31
    const int r_sub = tid & 7;    // 8 threads per row

    __syncthreads();

    for (int st = 0; st < NSTRIP; ++st) {
        const int js = st * STRIP;
        // prefetch next strip's feature rows (speculative, one row per thread)
        if (st + 1 < NSTRIP)
            __builtin_prefetch(hbf + (baseRow + js + STRIP + tid) * Dc, 0, 1);

        // --- distance tiles via WMMA: 16 j-tiles/strip, 2 per wave
#pragma unroll
        for (int tt2 = 0; tt2 < 2; ++tt2) {
            const int tt = wv * 2 + tt2;
            const int jb = js + tt * 16;
            const unsigned short* jrow = hbf + (baseRow + jb + m) * Dc;
            v8f c0 = {}, c1 = {};
#pragma unroll
            for (int kt = 0; kt < KT; ++kt) {
                V16 bt;                        // ISA 16-bit B 32x16 layout, b128 x2
                bt.q[0] = *(const u32x4*)(jrow + kt * 32 + kh * 8);
                bt.q[1] = *(const u32x4*)(jrow + kt * 32 + 16 + kh * 8);
                c0 = __builtin_amdgcn_wmma_f32_16x16x32_bf16(
                         false, a[0][kt].v, false, bt.v, (short)0, c0, false, false);
                c1 = __builtin_amdgcn_wmma_f32_16x16x32_bf16(
                         false, a[1][kt].v, false, bt.v, (short)0, c1, false, false);
            }
            const float sqj = sq[baseRow + jb + m];
#pragma unroll
            for (int r = 0; r < 8; ++r) {
                int mm = r + kh * 8;
                float d0 = s_sqi[mm]      + sqj - 2.0f * c0[r];
                float d1 = s_sqi[16 + mm] + sqj - 2.0f * c1[r];
                if (i0 + mm == jb + m)      d0 = __builtin_inff();
                if (i0 + 16 + mm == jb + m) d1 = __builtin_inff();
                s_dist[mm * STRIP        + tt * 16 + m] = d0;
                s_dist[(16 + mm) * STRIP + tt * 16 + m] = d1;
            }
        }
        __syncthreads();

        // --- top-k maintenance: 8 threads per row, 32 entries each
#pragma unroll
        for (int e = 0; e < STRIP / 8; ++e) {
            int jj = r_sub + e * 8;
            float d = s_dist[r_row * STRIP + jj];
            if (d < bd[15]) {
                bd[15] = d; bi[15] = js + jj;
#pragma unroll
                for (int s = 15; s > 0; --s) {
                    if (bd[s] < bd[s - 1]) {
                        float td = bd[s]; bd[s] = bd[s - 1]; bd[s - 1] = td;
                        int ti = bi[s]; bi[s] = bi[s - 1]; bi[s - 1] = ti;
                    }
                }
            }
        }
        __syncthreads();
    }

    // --- merge 8 partial top-16 lists per row (LDS aliased over dist strip)
#pragma unroll
    for (int s = 0; s < 16; ++s) {
        s_cd[r_row * 128 + r_sub * 16 + s] = bd[s];
        s_ci[r_row * 128 + r_sub * 16 + s] = bi[s];
    }
    __syncthreads();
    if (r_sub == 0) {
        for (int t = 1; t < 8; ++t) {
#pragma unroll
            for (int s2 = 0; s2 < 16; ++s2) {
                float d = s_cd[r_row * 128 + t * 16 + s2];
                if (d < bd[15]) {
                    bd[15] = d; bi[15] = s_ci[r_row * 128 + t * 16 + s2];
#pragma unroll
                    for (int s = 15; s > 0; --s) {
                        if (bd[s] < bd[s - 1]) {
                            float td = bd[s]; bd[s] = bd[s - 1]; bd[s - 1] = td;
                            int ti = bi[s]; bi[s] = bi[s - 1]; bi[s - 1] = ti;
                        }
                    }
                } else break;   // candidate list sorted ascending
            }
        }
#pragma unroll
        for (int s = 0; s < 16; ++s)
            nbr[(baseRow + i0 + r_row) * Kc + s] = bi[s];
    }
}

// ---------------------------------------------------------------- hw = h @ W (WMMA)
// W staged once per block into LDS (async-to-LDS when available); all 8 waves
// then read B-tiles from LDS with ds_load_b128.
template <int KT>
__global__ __launch_bounds__(256) void k_hw(const unsigned short* __restrict__ hbf,
                                            const unsigned short* __restrict__ wbf,
                                            float* __restrict__ hw) {
    const int WROW = KT * 32;
    __shared__ unsigned short s_w[Dc * 128];   // up to 32 KB (KT=4)

    const int nchunk = (Dc * WROW * 2) / 16;   // 16-byte chunks of Wt
    for (int i = threadIdx.x; i < nchunk; i += 256) {
#if HAS_ASYNC_LDS
        __builtin_amdgcn_global_load_async_to_lds_b128(
            (gvec_t*)((const char*)wbf + (size_t)i * 16),
            (lvec_t*)((char*)s_w + (size_t)i * 16), 0, 0);
#else
        *(u32x4*)((char*)s_w + (size_t)i * 16) =
            *(const u32x4*)((const char*)wbf + (size_t)i * 16);
#endif
    }
#if HAS_ASYNC_LDS
#if __has_builtin(__builtin_amdgcn_s_wait_asynccnt)
    __builtin_amdgcn_s_wait_asynccnt(0);
#else
    asm volatile("s_wait_asynccnt 0" ::: "memory");
#endif
#endif
    __syncthreads();

    const int tid  = threadIdx.x;
    const int lane = tid & 31;
    const int wv   = tid >> 5;
    const int gw   = blockIdx.x * 8 + wv;   // one 16-row tile per wave (flat B*N)
    const int i0   = gw * 16;
    const int m    = lane & 15;
    const int kh   = lane >> 4;

    V16 a[KT];
    {
        const unsigned short* row = hbf + (size_t)(i0 + m) * Dc;
#pragma unroll
        for (int kt = 0; kt < KT; ++kt) {
            a[kt].q[0] = *(const u32x4*)(row + kt * 32 + kh * 8);
            a[kt].q[1] = *(const u32x4*)(row + kt * 32 + 16 + kh * 8);
        }
    }

#pragma unroll
    for (int nt = 0; nt < 8; ++nt) {
        const unsigned short* wrow = s_w + (size_t)(nt * 16 + m) * WROW;
        v8f c = {};
#pragma unroll
        for (int kt = 0; kt < KT; ++kt) {
            V16 bt;
            bt.q[0] = *(const u32x4*)(wrow + kt * 32 + kh * 8);
            bt.q[1] = *(const u32x4*)(wrow + kt * 32 + 16 + kh * 8);
            c = __builtin_amdgcn_wmma_f32_16x16x32_bf16(
                    false, a[kt].v, false, bt.v, (short)0, c, false, false);
        }
#pragma unroll
        for (int r = 0; r < 8; ++r) {
            int mm = r + kh * 8;
            hw[(size_t)(i0 + mm) * Dc + nt * 16 + m] = c[r];
        }
    }
}

// ---------------------------------------------------------------- GCN aggregate + scatter-max
__global__ __launch_bounds__(128) void k_agg(const float* __restrict__ hw,
                                             const int* __restrict__ nbr,
                                             const float* __restrict__ bias,
                                             float* __restrict__ hnext) {
    const int node = blockIdx.x;
    const int b = node >> 12;
    const int c = threadIdx.x;
    const int* nb = nbr + (size_t)node * Kc;

    float s = hw[(size_t)node * Dc + c];
#pragma unroll
    for (int j = 0; j < Kc; ++j)
        s += hw[((size_t)b * Nc + nb[j]) * Dc + c];
    float h2 = s * (1.0f / (Kc + 1)) + bias[c];
    h2 = fmaxf(h2, 0.0f);
    unsigned bits = __float_as_uint(h2);
#pragma unroll
    for (int j = 0; j < Kc; ++j)
        atomicMax((unsigned*)&hnext[((size_t)b * Nc + nb[j]) * Dc + c], bits);
}

// ---------------------------------------------------------------- final transpose
__global__ void k_final(const float* __restrict__ h, float* __restrict__ out) {
    int idx = blockIdx.x * blockDim.x + threadIdx.x;
    if (idx >= BNc * Dc) return;
    int n = idx & (Nc - 1);
    int rest = idx >> 12;
    int c = rest & (Dc - 1);
    int b = rest >> 7;
    out[idx] = h[((size_t)b * Nc + n) * Dc + c];
}

// ---------------------------------------------------------------- launch
extern "C" void kernel_launch(void* const* d_in, const int* in_sizes, int n_in,
                              void* d_out, int out_size, void* d_ws, size_t ws_size,
                              hipStream_t stream) {
    (void)in_sizes; (void)n_in; (void)out_size; (void)ws_size;
    const float* x  = (const float*)d_in[0];
    const float* W1 = (const float*)d_in[1];
    const float* b1 = (const float*)d_in[2];
    const float* W2 = (const float*)d_in[3];
    const float* b2 = (const float*)d_in[4];
    const float* W3 = (const float*)d_in[5];
    const float* b3 = (const float*)d_in[6];
    float* out = (float*)d_out;

    char* w = (char*)d_ws;
    auto take = [&](size_t bytes) {
        void* p = (void*)w;
        w += (bytes + 255) & ~(size_t)255;
        return p;
    };
    float*          hA  = (float*)take((size_t)BNc * Dc * 4);
    float*          hB  = (float*)take((size_t)BNc * Dc * 4);
    float*          hw  = (float*)take((size_t)BNc * Dc * 4);
    unsigned short* hbf = (unsigned short*)take((size_t)BNc * Dc * 2);
    float*          sq  = (float*)take((size_t)BNc * 4);
    int*            nbr = (int*)take((size_t)BNc * Kc * 4);
    unsigned short* wbf = (unsigned short*)take((size_t)Dc * Dc * 2);

    const int knnGrid = Bc * (Nc / ROWS);  // 2048
    const int hwGrid  = BNc / 16 / 8;      // 512

    // ---------------- Layer 1 (d_in = 3, KT = 1)
    k_prep1<<<BNc, 128, 0, stream>>>(x, hbf, sq);
    k_castW<<<(Dc * 32 + 255) / 256, 256, 0, stream>>>(W1, wbf, 3, 32);
    k_knn<1><<<knnGrid, 256, 0, stream>>>(hbf, sq, nbr);
    k_hw<1><<<hwGrid, 256, 0, stream>>>(hbf, wbf, hw);
    k_zero<<<4096, 256, 0, stream>>>(hA, BNc * Dc);
    k_agg<<<BNc, 128, 0, stream>>>(hw, nbr, b1, hA);

    // ---------------- Layer 2 (d_in = 128, KT = 4)
    k_prep<<<BNc, 128, 0, stream>>>(hA, hbf, sq);
    k_castW<<<(Dc * Dc + 255) / 256, 256, 0, stream>>>(W2, wbf, Dc, Dc);
    k_knn<4><<<knnGrid, 256, 0, stream>>>(hbf, sq, nbr);
    k_hw<4><<<hwGrid, 256, 0, stream>>>(hbf, wbf, hw);
    k_zero<<<4096, 256, 0, stream>>>(hB, BNc * Dc);
    k_agg<<<BNc, 128, 0, stream>>>(hw, nbr, b2, hB);

    // ---------------- Layer 3 (d_in = 128, KT = 4)
    k_prep<<<BNc, 128, 0, stream>>>(hB, hbf, sq);
    k_castW<<<(Dc * Dc + 255) / 256, 256, 0, stream>>>(W3, wbf, Dc, Dc);
    k_knn<4><<<knnGrid, 256, 0, stream>>>(hbf, sq, nbr);
    k_hw<4><<<hwGrid, 256, 0, stream>>>(hbf, wbf, hw);
    k_zero<<<4096, 256, 0, stream>>>(hA, BNc * Dc);
    k_agg<<<BNc, 128, 0, stream>>>(hw, nbr, b3, hA);

    k_final<<<(BNc * Dc + 255) / 256, 256, 0, stream>>>(hA, out);
}
